// GNN_8263517077773
// MI455X (gfx1250) — matrix operand order repaired and checked by reading, not compiled, not run on hardware
//
#include <hip/hip_runtime.h>

typedef __attribute__((ext_vector_type(16))) _Float16 v16h;
typedef __attribute__((ext_vector_type(8)))  float    v8f;

#define B_  128
#define N_  64
#define FI_ 8
#define F_  32
#define NN_ 64
#define E_  4032

// fragment table indices (each fragment = 32 lanes x 16 halves = 1KB)
#define FR_W2A 0    // 3 kt x 4 nt  (K=96 pad of 66, N=64)
#define FR_W2B 12   // 2 kt x 2 nt  (K=64, N=32)
#define FR_W2C 16   // 1 kt x 2 nt  (K=32, N=32)
#define FR_WX  18   // 1 kt x 12 nt (K=32, N=192)
#define FR_UX  30   // 2 kt x 12 nt (K=64, N=192)
#define FR_W3B 54   // 2 kt x 2 nt  (K=64, N=32)
#define FR_W4A 58   // 1 kt x 4 nt  (K=32, N=64)
#define FR_W4B 62   // 2 kt x 2 nt  (K=64, N=32)
#define FR_W4C 66   // 1 kt x 2 nt  (K=32, N=32)
#define NFRAG  68

#define WMMA(a, b, c) \
  __builtin_amdgcn_wmma_f32_16x16x32_f16(false, (a), false, (b), (short)0, (c), false, false)

union HU {
  v16h v;
  unsigned int u[8];
  _Float16 h[16];
};

__device__ __forceinline__ float sigm_f(float x) { return 1.0f / (1.0f + __expf(-x)); }

// jax.nn.gelu (approximate=True, tanh form), rewritten branch-free:
//   0.5*x*(1+tanh(u)) == x * sigmoid(2*u),  u = 0.79788456*(x + 0.044715*x^3)
__device__ __forceinline__ float gelu_f(float x) {
  float x3 = x * x * x;
  return x * sigm_f(1.5957691216057308f * (x + 0.044715f * x3));
}

__device__ __forceinline__ v16h load_frag(const _Float16* frags, int f, int lane) {
  return ((const v16h*)frags)[f * 32 + lane];
}

// Read a 16x32 f16 A-fragment (ISA A-layout) out of a 16-row LDS tile.
__device__ __forceinline__ v16h lds_loadA(const _Float16* base16rows, int ldRow, int kt, int lane) {
  int m = lane & 15, khi = lane >> 4;
  const unsigned int* rp = (const unsigned int*)(base16rows + m * ldRow);
  HU r;
#pragma unroll
  for (int v = 0; v < 8; ++v) {
    int pair = (v < 4) ? (4 * khi + v) : (4 * khi + v + 4);
    r.u[v] = rp[kt * 16 + pair];
  }
  return r.v;
}

// ---------------------------------------------------------------------------
// Weight swizzle: f32 [K,N] row-major -> f16 B-fragments (32x16 per 16x16x32).
// B-layout: lane = nt*16 + (lane&15) column; half slot s=2v+p -> K = kt*32 + (lane>>4)*8 + s + (s>=8 ? 8 : 0)
// ---------------------------------------------------------------------------
__global__ void prep_weights(const float* W2a, const float* W2b, const float* W2c,
                             const float* Wx, const float* Ux, const float* W3b,
                             const float* W4a, const float* W4b, const float* W4c,
                             _Float16* frags) {
  int f = blockIdx.x;        // fragment index
  int t = threadIdx.x;       // 0..255
  int lane = t >> 3;         // 0..31
  int v = t & 7;             // 0..7
  const float* src;
  int Ksrc, Ncols, kt, nt;
  if (f < 12)      { src = W2a; Ksrc = 66; Ncols = 64;  kt = f / 4;        nt = f % 4; }
  else if (f < 16) { src = W2b; Ksrc = 64; Ncols = 32;  kt = (f - 12) / 2; nt = (f - 12) % 2; }
  else if (f < 18) { src = W2c; Ksrc = 32; Ncols = 32;  kt = 0;            nt = f - 16; }
  else if (f < 30) { src = Wx;  Ksrc = 32; Ncols = 192; kt = 0;            nt = f - 18; }
  else if (f < 54) { src = Ux;  Ksrc = 64; Ncols = 192; kt = (f - 30) / 12; nt = (f - 30) % 12; }
  else if (f < 58) { src = W3b; Ksrc = 64; Ncols = 32;  kt = (f - 54) / 2; nt = (f - 54) % 2; }
  else if (f < 62) { src = W4a; Ksrc = 32; Ncols = 64;  kt = 0;            nt = f - 58; }
  else if (f < 66) { src = W4b; Ksrc = 64; Ncols = 32;  kt = (f - 62) / 2; nt = (f - 62) % 2; }
  else             { src = W4c; Ksrc = 32; Ncols = 32;  kt = 0;            nt = f - 66; }
  int n = nt * 16 + (lane & 15);
  int khi = lane >> 4;
#pragma unroll
  for (int p = 0; p < 2; ++p) {
    int K = kt * 32 + khi * 8 + 2 * v + p + (v >= 4 ? 8 : 0);
    float val = (K < Ksrc) ? src[K * Ncols + n] : 0.0f;
    frags[(f * 32 + lane) * 16 + 2 * v + p] = (_Float16)val;
  }
}

// ---------------------------------------------------------------------------
// Init: node = node0 @ W1 + b1 (stored f16), h copied to f32 + f16 workspace.
// ---------------------------------------------------------------------------
__global__ void init_kernel(const float* node0, const float* W1, const float* b1,
                            const float* hIn, _Float16* nodeF16, float* hF32, _Float16* hF16) {
  int idx = blockIdx.x * 256 + threadIdx.x;
  if (idx < B_ * N_ * NN_) {
    float v = hIn[idx];
    hF32[idx] = v;
    hF16[idx] = (_Float16)v;
  }
  if (idx < B_ * N_ * F_) {
    int g = idx / F_;
    int c = idx % F_;
    float s = b1[c];
#pragma unroll
    for (int k = 0; k < FI_; ++k) s += node0[g * FI_ + k] * W1[k * F_ + c];
    nodeF16[idx] = (_Float16)s;
  }
}

// ---------------------------------------------------------------------------
// Fused edge MLP + segment sum. Grid (N, B); 4 waves, each owns 16 edges.
// ---------------------------------------------------------------------------
__global__ __launch_bounds__(128) void edge_mlp(const _Float16* __restrict__ nodeF16,
                                                const float* __restrict__ edge,
                                                const _Float16* __restrict__ frags,
                                                const float* __restrict__ b2a,
                                                const float* __restrict__ b2b,
                                                const float* __restrict__ b2c,
                                                float* __restrict__ m_i) {
  __shared__ _Float16 ldsM1[4][16][64];
  __shared__ _Float16 ldsM2[4][16][32];
  __shared__ float ldsSum[32];

  int i = blockIdx.x, b = blockIdx.y;
  int t = threadIdx.x, w = t >> 5, lane = t & 31;
  int m = lane & 15, khi = lane >> 4;
  __builtin_prefetch(frags, 0, 3);
  if (t < 32) ldsSum[t] = 0.0f;
  __syncthreads();

  // ---- build A fragments (16 rows = 16 edges of this wave) ----
  int local = w * 16 + m;
  bool valid = local < 63;
  int lcl = valid ? local : 62;
  int j = lcl + (lcl >= i ? 1 : 0);
  const unsigned int* ni = (const unsigned int*)(nodeF16 + (b * N_ + i) * F_);
  const unsigned int* nj = (const unsigned int*)(nodeF16 + (b * N_ + j) * F_);
  HU a0, a1, a2;
#pragma unroll
  for (int v = 0; v < 8; ++v) {
    int pair = (v < 4) ? (4 * khi + v) : (4 * khi + v + 4);
    a0.u[v] = ni[pair];
    a1.u[v] = valid ? nj[pair] : 0u;
    a2.u[v] = 0u;
  }
  {  // edge features sit at K=64,65 -> kt=2, slot0/1, lanes khi==0 only
    int e2 = (b * E_ + i * 63 + lcl) * 2;
    float e0 = (khi == 0 && valid) ? edge[e2 + 0] : 0.0f;
    float e1 = (khi == 0 && valid) ? edge[e2 + 1] : 0.0f;
    HU pk;
    pk.h[0] = (_Float16)e0;
    pk.h[1] = (_Float16)e1;
    if (khi == 0) a2.u[0] = pk.u[0];
  }

  // ---- layer 1: [16,96] x [96,64] -> gelu -> ldsM1 ----
#pragma unroll
  for (int nt = 0; nt < 4; ++nt) {
    v8f acc = {};
    acc = WMMA(a0.v, load_frag(frags, FR_W2A + 0 * 4 + nt, lane), acc);
    acc = WMMA(a1.v, load_frag(frags, FR_W2A + 1 * 4 + nt, lane), acc);
    acc = WMMA(a2.v, load_frag(frags, FR_W2A + 2 * 4 + nt, lane), acc);
    float bias = b2a[nt * 16 + m];
    int c = nt * 16 + m;
#pragma unroll
    for (int v = 0; v < 8; ++v)
      ldsM1[w][v + 8 * khi][c] = (_Float16)gelu_f(acc[v] + bias);
  }
  __syncthreads();

  // ---- layer 2: [16,64] x [64,32] -> gelu -> ldsM2 ----
  v16h a20 = lds_loadA(&ldsM1[w][0][0], 64, 0, lane);
  v16h a21 = lds_loadA(&ldsM1[w][0][0], 64, 1, lane);
#pragma unroll
  for (int nt = 0; nt < 2; ++nt) {
    v8f acc = {};
    acc = WMMA(a20, load_frag(frags, FR_W2B + 0 * 2 + nt, lane), acc);
    acc = WMMA(a21, load_frag(frags, FR_W2B + 1 * 2 + nt, lane), acc);
    float bias = b2b[nt * 16 + m];
    int c = nt * 16 + m;
#pragma unroll
    for (int v = 0; v < 8; ++v)
      ldsM2[w][v + 8 * khi][c] = (_Float16)gelu_f(acc[v] + bias);
  }
  __syncthreads();

  // ---- layer 3: [16,32] x [32,32] -> gelu -> masked row-sum -> ldsSum ----
  v16h a3 = lds_loadA(&ldsM2[w][0][0], 32, 0, lane);
#pragma unroll
  for (int nt = 0; nt < 2; ++nt) {
    v8f acc = {};
    acc = WMMA(a3, load_frag(frags, FR_W2C + nt, lane), acc);
    float bias = b2c[nt * 16 + m];
    float s = 0.0f;
#pragma unroll
    for (int v = 0; v < 8; ++v) {
      int row = w * 16 + v + 8 * khi;  // edge-local index within the 63 edges
      float val = gelu_f(acc[v] + bias);
      if (row < 63) s += val;
    }
    atomicAdd(&ldsSum[nt * 16 + m], s);
  }
  __syncthreads();
  if (t < 32) m_i[(b * N_ + i) * F_ + t] = ldsSum[t];
}

// ---------------------------------------------------------------------------
// GRU cell + fc3b. 1 wave per 16-row tile of the flat [B*N] rows.
// ---------------------------------------------------------------------------
__global__ __launch_bounds__(128) void gru_node(const float* __restrict__ m_i,
                                                float* __restrict__ hF32,
                                                _Float16* __restrict__ hF16,
                                                const _Float16* __restrict__ frags,
                                                const float* __restrict__ bx,
                                                const float* __restrict__ bh,
                                                const float* __restrict__ b3b,
                                                _Float16* __restrict__ nodeF16) {
  __shared__ _Float16 ldsH[4][16][64];
  int t = threadIdx.x, w = t >> 5, lane = t & 31;
  int m = lane & 15, khi = lane >> 4;
  int tile = blockIdx.x * 4 + w;  // 0..511
  int gm = tile * 16 + m;

  // A fragment for m_i (f32 -> f16, A-layout), K=32
  HU am;
  const float2* mr = (const float2*)(m_i + gm * F_);
  // A fragments for h (f16), K=64 -> kt 0,1
  const unsigned int* hr = (const unsigned int*)(hF16 + gm * NN_);
  HU ah0, ah1;
#pragma unroll
  for (int v = 0; v < 8; ++v) {
    int pair = (v < 4) ? (4 * khi + v) : (4 * khi + v + 4);
    float2 pr = mr[pair];
    am.h[2 * v] = (_Float16)pr.x;
    am.h[2 * v + 1] = (_Float16)pr.y;
    ah0.u[v] = hr[pair];
    ah1.u[v] = hr[16 + pair];
  }

#pragma unroll
  for (int nt = 0; nt < 4; ++nt) {
    v8f xr = {}, xz = {}, xn = {}, hrv = {}, hzv = {}, hnv = {};
    xr = WMMA(am.v, load_frag(frags, FR_WX + nt, lane), xr);
    xz = WMMA(am.v, load_frag(frags, FR_WX + 4 + nt, lane), xz);
    xn = WMMA(am.v, load_frag(frags, FR_WX + 8 + nt, lane), xn);
    hrv = WMMA(ah0.v, load_frag(frags, FR_UX + nt, lane), hrv);
    hrv = WMMA(ah1.v, load_frag(frags, FR_UX + 12 + nt, lane), hrv);
    hzv = WMMA(ah0.v, load_frag(frags, FR_UX + 4 + nt, lane), hzv);
    hzv = WMMA(ah1.v, load_frag(frags, FR_UX + 12 + 4 + nt, lane), hzv);
    hnv = WMMA(ah0.v, load_frag(frags, FR_UX + 8 + nt, lane), hnv);
    hnv = WMMA(ah1.v, load_frag(frags, FR_UX + 12 + 8 + nt, lane), hnv);
    int c = nt * 16 + m;
    float bxr = bx[c], bxz = bx[64 + c], bxn = bx[128 + c];
    float bhr = bh[c], bhz = bh[64 + c], bhn = bh[128 + c];
#pragma unroll
    for (int v = 0; v < 8; ++v) {
      int g = tile * 16 + v + 8 * khi;
      float hOld = hF32[g * NN_ + c];
      float r = sigm_f(xr[v] + bxr + hrv[v] + bhr);
      float z = sigm_f(xz[v] + bxz + hzv[v] + bhz);
      float n = gelu_f(xn[v] + bxn + r * (hnv[v] + bhn));
      float hNew = (1.0f - z) * n + z * hOld;
      hF32[g * NN_ + c] = hNew;
      hF16[g * NN_ + c] = (_Float16)hNew;
      ldsH[w][v + 8 * khi][c] = (_Float16)hNew;
    }
  }
  __syncthreads();

  // fc3b: node = h_new @ W3b + b3b  ([16,64] x [64,32])
  v16h n0 = lds_loadA(&ldsH[w][0][0], 64, 0, lane);
  v16h n1 = lds_loadA(&ldsH[w][0][0], 64, 1, lane);
#pragma unroll
  for (int nt = 0; nt < 2; ++nt) {
    v8f acc = {};
    acc = WMMA(n0, load_frag(frags, FR_W3B + 0 * 2 + nt, lane), acc);
    acc = WMMA(n1, load_frag(frags, FR_W3B + 1 * 2 + nt, lane), acc);
    float bias = b3b[nt * 16 + m];
    int c = nt * 16 + m;
#pragma unroll
    for (int v = 0; v < 8; ++v) {
      int g = tile * 16 + v + 8 * khi;
      nodeF16[g * F_ + c] = (_Float16)(acc[v] + bias);
    }
  }
}

// ---------------------------------------------------------------------------
// Readout: ((node @ W4a + b4a) @ W4b + b4b) @ W4c + b4c  (no activations)
// ---------------------------------------------------------------------------
__global__ __launch_bounds__(128) void readout(const _Float16* __restrict__ nodeF16,
                                               const _Float16* __restrict__ frags,
                                               const float* __restrict__ b4a,
                                               const float* __restrict__ b4b,
                                               const float* __restrict__ b4c,
                                               float* __restrict__ out) {
  __shared__ _Float16 ldsA[4][16][64];
  __shared__ _Float16 ldsB[4][16][32];
  int t = threadIdx.x, w = t >> 5, lane = t & 31;
  int m = lane & 15, khi = lane >> 4;
  int tile = blockIdx.x * 4 + w;
  int gm = tile * 16 + m;

  const unsigned int* nr = (const unsigned int*)(nodeF16 + gm * F_);
  HU an;
#pragma unroll
  for (int v = 0; v < 8; ++v) {
    int pair = (v < 4) ? (4 * khi + v) : (4 * khi + v + 4);
    an.u[v] = nr[pair];
  }

#pragma unroll
  for (int nt = 0; nt < 4; ++nt) {
    v8f acc = {};
    acc = WMMA(an.v, load_frag(frags, FR_W4A + nt, lane), acc);
    float bias = b4a[nt * 16 + m];
    int c = nt * 16 + m;
#pragma unroll
    for (int v = 0; v < 8; ++v)
      ldsA[w][v + 8 * khi][c] = (_Float16)(acc[v] + bias);
  }
  __syncthreads();

  v16h a0 = lds_loadA(&ldsA[w][0][0], 64, 0, lane);
  v16h a1 = lds_loadA(&ldsA[w][0][0], 64, 1, lane);
#pragma unroll
  for (int nt = 0; nt < 2; ++nt) {
    v8f acc = {};
    acc = WMMA(a0, load_frag(frags, FR_W4B + 0 * 2 + nt, lane), acc);
    acc = WMMA(a1, load_frag(frags, FR_W4B + 1 * 2 + nt, lane), acc);
    float bias = b4b[nt * 16 + m];
    int c = nt * 16 + m;
#pragma unroll
    for (int v = 0; v < 8; ++v)
      ldsB[w][v + 8 * khi][c] = (_Float16)(acc[v] + bias);
  }
  __syncthreads();

  v16h a2 = lds_loadA(&ldsB[w][0][0], 32, 0, lane);
#pragma unroll
  for (int nt = 0; nt < 2; ++nt) {
    v8f acc = {};
    acc = WMMA(a2, load_frag(frags, FR_W4C + nt, lane), acc);
    float bias = b4c[nt * 16 + m];
    int c = nt * 16 + m;
#pragma unroll
    for (int v = 0; v < 8; ++v) {
      int g = tile * 16 + v + 8 * khi;
      out[g * F_ + c] = acc[v] + bias;
    }
  }
}

// ---------------------------------------------------------------------------
extern "C" void kernel_launch(void* const* d_in, const int* in_sizes, int n_in,
                              void* d_out, int out_size, void* d_ws, size_t ws_size,
                              hipStream_t stream) {
  const float* node0 = (const float*)d_in[0];
  const float* edge  = (const float*)d_in[1];
  // d_in[2], d_in[3]: node_mean / node_vari (unused by reference)
  const float* hIn   = (const float*)d_in[4];
  const float* W1    = (const float*)d_in[5];
  const float* b1    = (const float*)d_in[6];
  const float* W2a   = (const float*)d_in[7];
  const float* b2a   = (const float*)d_in[8];
  const float* W2b   = (const float*)d_in[9];
  const float* b2b   = (const float*)d_in[10];
  const float* W2c   = (const float*)d_in[11];
  const float* b2c   = (const float*)d_in[12];
  const float* Wx    = (const float*)d_in[13];
  const float* Ux    = (const float*)d_in[14];
  const float* bx    = (const float*)d_in[15];
  const float* bh    = (const float*)d_in[16];
  const float* W3b   = (const float*)d_in[17];
  const float* b3b   = (const float*)d_in[18];
  const float* W4a   = (const float*)d_in[19];
  const float* b4a   = (const float*)d_in[20];
  const float* W4b   = (const float*)d_in[21];
  const float* b4b   = (const float*)d_in[22];
  const float* W4c   = (const float*)d_in[23];
  const float* b4c   = (const float*)d_in[24];

  char* ws = (char*)d_ws;
  _Float16* wsFrag  = (_Float16*)(ws);                 //  68 KB  (swizzled weights)
  _Float16* nodeF16 = (_Float16*)(ws + 69632);         // 512 KB  [B,N,F] f16
  float*    hF32    = (float*)(ws + 593920);           //   2 MB  [B,N,NN] f32
  _Float16* hF16    = (_Float16*)(ws + 2691072);       //   1 MB  [B,N,NN] f16
  float*    m_i     = (float*)(ws + 3739648);          //   1 MB  [B,N,F] f32

  prep_weights<<<NFRAG, 256, 0, stream>>>(W2a, W2b, W2c, Wx, Ux, W3b, W4a, W4b, W4c, wsFrag);
  init_kernel<<<2048, 256, 0, stream>>>(node0, W1, b1, hIn, nodeF16, hF32, hF16);
  for (int it = 0; it < 3; ++it) {
    edge_mlp<<<dim3(N_, B_), 128, 0, stream>>>(nodeF16, edge, wsFrag, b2a, b2b, b2c, m_i);
    gru_node<<<128, 128, 0, stream>>>(m_i, hF32, hF16, wsFrag, bx, bh, b3b, nodeF16);
  }
  readout<<<128, 128, 0, stream>>>(nodeF16, wsFrag, b4a, b4b, b4c, (float*)d_out);
}
